// MSDeformAttn_64433099375176
// MI455X (gfx1250) — matrix operand order repaired and checked by reading, not compiled, not run on hardware
//
#include <hip/hip_runtime.h>
#include <hip/hip_bf16.h>

typedef __attribute__((ext_vector_type(16))) _Float16 v16h;
typedef __attribute__((ext_vector_type(8)))  _Float16 v8h;
typedef __attribute__((ext_vector_type(8)))  float    v8f;
typedef __attribute__((ext_vector_type(4)))  float    v4f;

#define D_MODEL   256
#define N_HEADS   8
#define DH        32
#define N_LEVELS  4
#define N_POINTS  4
#define BATCH     8
#define LEN_Q     8192
#define LEN_IN    19560   // 92*160 + 46*80 + 23*40 + 12*20

// ---------------------------------------------------------------------------
// Streaming fp32 -> f16 conversion, 8 elements per thread-iteration.
// ---------------------------------------------------------------------------
__global__ void __launch_bounds__(256)
cvt_f32_f16_kernel(const float* __restrict__ in, _Float16* __restrict__ out, int n8)
{
    for (int i = blockIdx.x * blockDim.x + threadIdx.x; i < n8;
         i += gridDim.x * blockDim.x) {
        const v4f* p = (const v4f*)in + (size_t)i * 2;
        v4f a = p[0], b = p[1];
        v8h o;
#pragma unroll
        for (int r = 0; r < 4; ++r) { o[r] = (_Float16)a[r]; o[4 + r] = (_Float16)b[r]; }
        ((v8h*)out)[i] = o;
    }
}

// ---------------------------------------------------------------------------
// f16-WMMA GEMM:  out[M,N] = A[M,256] @ W[256,N] + bias[N]
// A: f16, read directly from global in fragment layout (2x b128 per fragment).
// W: fp32, converted + pre-swizzled once per block into LDS in exact
//    B-fragment order, kt-major so one k-step's fragments span only NT KB
//    (small DS offset immediates from a single base).
// 256 threads = 8 waves; a wave owns a 16-row strip, processed in strictly
// sequential passes of <=8 N-tiles to keep live VGPRs under 256.
// ---------------------------------------------------------------------------
template <int NT, bool OUT_HALF>
__global__ void __launch_bounds__(256)
gemm_wmma_kernel(const _Float16* __restrict__ Ah,
                 const float* __restrict__ Wg,
                 const float* __restrict__ bias,
                 float* __restrict__ outF,
                 _Float16* __restrict__ outH,
                 int M)
{
    constexpr int N      = NT * 16;
    constexpr int KT     = 8;                    // 256 / 32
    constexpr int NH     = (NT > 8) ? 8 : NT;    // N-tiles per pass
    constexpr int PASSES = NT / NH;

    extern __shared__ char smem[];
    _Float16* Wlds = (_Float16*)smem;            // N*256 halves, fragment order (kt-major)

    const int tid    = threadIdx.x;
    const int wave   = tid >> 5;
    const int lane   = tid & 31;
    const int lane15 = lane & 15;
    const int hi     = lane >> 4;                // K-half / row-half selector

    // Convert + swizzle W into kt-major fragment order; source side coalesced.
    // frag element j of lane (hi2,l15) of tile (kt,nt) = W[kt*32+hi2*16+j][nt*16+l15]
    for (int idx = tid; idx < 256 * N; idx += 256) {
        const int k     = idx / N;
        const int n     = idx - k * N;
        const int nt    = n >> 4, l15 = n & 15;
        const int kt    = k >> 5, kk  = k & 31;
        const int hi2   = kk >> 4, j  = kk & 15;
        const int lane2 = hi2 * 16 + l15;
        Wlds[(((kt * NT + nt) * 32 + lane2) << 4) + j] = (_Float16)Wg[idx];
    }
    __syncthreads();

    const int strips = M >> 4;
    for (int strip = blockIdx.x * 8 + wave; strip < strips; strip += gridDim.x * 8) {
        const int row0 = strip << 4;
        const _Float16* arow = Ah + (size_t)(row0 + lane15) * D_MODEL + hi * 16;

#pragma unroll 1   // passes MUST stay sequential: caps live accumulators at 64 VGPRs
        for (int pass = 0; pass < PASSES; ++pass) {
            const int nt0 = pass * NH;

            v8f acc[NH];
#pragma unroll
            for (int j = 0; j < NH; ++j) {
                const float bj = bias[(nt0 + j) * 16 + lane15];
                v8f a;
#pragma unroll
                for (int r = 0; r < 8; ++r) a[r] = bj;
                acc[j] = a;
            }

#pragma unroll
            for (int kt = 0; kt < KT; ++kt) {
                const _Float16* ap = arow + kt * 32;
                v8h lo = *(const v8h*)(ap);
                v8h hh = *(const v8h*)(ap + 8);
                v16h afrag = __builtin_shufflevector(lo, hh,
                                 0,1,2,3,4,5,6,7,8,9,10,11,12,13,14,15);
                const _Float16* wbase = Wlds + (((kt * NT + nt0) * 32 + lane) << 4);
#pragma unroll
                for (int j = 0; j < NH; ++j) {
                    const v8h* bp = (const v8h*)(wbase + (j << 9));   // +j*32 lanes*16 halves
                    v16h bfrag = __builtin_shufflevector(bp[0], bp[1],
                                     0,1,2,3,4,5,6,7,8,9,10,11,12,13,14,15);
                    acc[j] = __builtin_amdgcn_wmma_f32_16x16x32_f16(
                                 false, afrag, false, bfrag, (short)0, acc[j], false, false);
                }
            }

            // D element r of lane t -> row = row0 + (t>=16)*8 + r, col = nt*16 + (t&15)
#pragma unroll
            for (int j = 0; j < NH; ++j) {
                const int col = (nt0 + j) * 16 + lane15;
#pragma unroll
                for (int r = 0; r < 8; ++r) {
                    const int row = row0 + hi * 8 + r;
                    if (OUT_HALF) outH[(size_t)row * N + col] = (_Float16)acc[j][r];
                    else          outF[(size_t)row * N + col] = acc[j][r];
                }
            }
        }
    }
}

// ---------------------------------------------------------------------------
// Softmax over 16 (level,point) logits per (b,q,h) + location computation,
// both in place. One thread per (b,q,h).
// ---------------------------------------------------------------------------
__global__ void __launch_bounds__(256)
softmax_loc_kernel(float* __restrict__ attn,       // [B*LQ*H, 16]
                   float* __restrict__ loc,        // [B*LQ*H, 16, 2]
                   const float* __restrict__ ref)  // [B, LQ, L, 2]
{
    const int tid = blockIdx.x * 256 + threadIdx.x;   // < 524288

    float* ap = attn + (size_t)tid * 16;
    float e[16];
    float m = ap[0];
#pragma unroll
    for (int i = 1; i < 16; ++i) m = fmaxf(m, ap[i]);
    float s = 0.f;
#pragma unroll
    for (int i = 0; i < 16; ++i) { e[i] = __expf(ap[i] - m); s += e[i]; }
    const float inv = 1.f / s;
#pragma unroll
    for (int i = 0; i < 16; ++i) ap[i] = e[i] * inv;

    const int bq = tid >> 3;
    const float* rp = ref + (size_t)bq * (N_LEVELS * 2);
    float* lp = loc + (size_t)tid * 32;
    const float iWl[4] = {1.f/160.f, 1.f/80.f, 1.f/40.f, 1.f/20.f};
    const float iHl[4] = {1.f/92.f, 1.f/46.f, 1.f/23.f, 1.f/12.f};
#pragma unroll
    for (int l = 0; l < N_LEVELS; ++l) {
        const float rx = rp[l * 2], ry = rp[l * 2 + 1];
#pragma unroll
        for (int p = 0; p < N_POINTS; ++p) {
            const int o = (l * 4 + p) * 2;
            lp[o]     = rx + lp[o]     * iWl[l];
            lp[o + 1] = ry + lp[o + 1] * iHl[l];
        }
    }
}

// ---------------------------------------------------------------------------
// Bilinear sampling + attention-weighted accumulation.
// One wave32 per (b,q,h); lane = channel (Dh == 32). Uniform scalar
// addressing for weights/locations; f16 value tensor is L2-resident.
// ---------------------------------------------------------------------------
__global__ void __launch_bounds__(256)
sampling_kernel(const _Float16* __restrict__ vh,    // [B*LEN_IN, 256] f16
                const float* __restrict__ attn,     // [B*LQ*H, 16]
                const float* __restrict__ loc,      // [B*LQ*H, 16, 2]
                _Float16* __restrict__ samp)        // [B*LQ, 256] f16
{
    const int wave = threadIdx.x >> 5;
    const int lane = threadIdx.x & 31;
    const int u = __builtin_amdgcn_readfirstlane(blockIdx.x * 8 + wave);
    const int h  = u & 7;
    const int bq = u >> 3;
    const int b  = bq >> 13;                         // LQ = 2^13

    const float* ap = attn + (size_t)u * 16;
    const float* lp = loc  + (size_t)u * 32;
    const _Float16* vbase = vh + (size_t)b * LEN_IN * D_MODEL + h * DH + lane;

    const int Hs[4] = {92, 46, 23, 12};
    const int Ws[4] = {160, 80, 40, 20};
    const int St[4] = {0, 14720, 18400, 19320};

    float acc = 0.f;
#pragma unroll
    for (int l = 0; l < N_LEVELS; ++l) {
        const int Hc = Hs[l], Wc = Ws[l], S = St[l];
#pragma unroll
        for (int p = 0; p < N_POINTS; ++p) {
            const int ip = l * 4 + p;
            const float w  = ap[ip];
            const float xs = lp[2 * ip]     * (float)Wc - 0.5f;
            const float ys = lp[2 * ip + 1] * (float)Hc - 0.5f;
            const float x0f = floorf(xs), y0f = floorf(ys);
            const float wx = xs - x0f, wy = ys - y0f;
            const int x0 = (int)x0f, y0 = (int)y0f;
            const int x1 = x0 + 1,   y1 = y0 + 1;

            float c00 = 0.f, c10 = 0.f, c01 = 0.f, c11 = 0.f;
            const bool vx0 = (x0 >= 0) & (x0 < Wc);
            const bool vx1 = (x1 >= 0) & (x1 < Wc);
            const bool vy0 = (y0 >= 0) & (y0 < Hc);
            const bool vy1 = (y1 >= 0) & (y1 < Hc);
            if (vx0 & vy0) c00 = (float)vbase[(size_t)(S + y0 * Wc + x0) * D_MODEL];
            if (vx1 & vy0) c10 = (float)vbase[(size_t)(S + y0 * Wc + x1) * D_MODEL];
            if (vx0 & vy1) c01 = (float)vbase[(size_t)(S + y1 * Wc + x0) * D_MODEL];
            if (vx1 & vy1) c11 = (float)vbase[(size_t)(S + y1 * Wc + x1) * D_MODEL];

            const float sv = c00 * (1.f - wx) * (1.f - wy)
                           + c10 * wx * (1.f - wy)
                           + c01 * (1.f - wx) * wy
                           + c11 * wx * wy;
            acc += w * sv;
        }
    }
    samp[(size_t)u * DH + lane] = (_Float16)acc;
}

// ---------------------------------------------------------------------------
// Launcher
// ---------------------------------------------------------------------------
extern "C" void kernel_launch(void* const* d_in, const int* in_sizes, int n_in,
                              void* d_out, int out_size, void* d_ws, size_t ws_size,
                              hipStream_t stream) {
    (void)in_sizes; (void)n_in; (void)out_size; (void)ws_size;

    const float* query  = (const float*)d_in[0];
    const float* refpt  = (const float*)d_in[1];
    const float* inputf = (const float*)d_in[2];
    const float* W_off  = (const float*)d_in[5];
    const float* b_off  = (const float*)d_in[6];
    const float* W_attn = (const float*)d_in[7];
    const float* b_attn = (const float*)d_in[8];
    const float* W_val  = (const float*)d_in[9];
    const float* b_val  = (const float*)d_in[10];
    const float* W_out  = (const float*)d_in[11];
    const float* b_out  = (const float*)d_in[12];
    float* out = (float*)d_out;

    // workspace layout (256B-aligned). samp_h aliases input_h (disjoint lifetimes:
    // input_h is consumed by GEMM-1, samp_h is produced by the sampling kernel later).
    char* ws = (char*)d_ws;
    _Float16* value_h = (_Float16*)ws;                         // 80,117,760 B
    float*    attn    = (float*)(ws + 80117760);               // 33,554,432 B
    float*    loc     = (float*)(ws + 113672192);              // 67,108,864 B
    _Float16* input_h = (_Float16*)(ws + 180781056);           // 80,117,760 B
    _Float16* samp_h  = input_h;                               // alias (33,554,432 B)
    _Float16* query_h = (_Float16*)(ws + 260898816);           // 33,554,432 B

    const int M_val = BATCH * LEN_IN;   // 156480
    const int M_q   = BATCH * LEN_Q;    // 65536

    const size_t lds_n256 = 256 * 256 * 2;  // 128 KB
    const size_t lds_n128 = 256 * 128 * 2;  //  64 KB

    // 0) activation conversions to f16 (streaming, HBM-bound)
    cvt_f32_f16_kernel<<<4096, 256, 0, stream>>>(inputf, input_h, M_val * D_MODEL / 8);
    cvt_f32_f16_kernel<<<4096, 256, 0, stream>>>(query,  query_h, M_q  * D_MODEL / 8);

    // 1) value projection: f16 in, f16 out
    gemm_wmma_kernel<16, true><<<512, 256, lds_n256, stream>>>(
        input_h, W_val, b_val, nullptr, value_h, M_val);

    // 2) attention logits (N=128): f16 in, f32 out
    gemm_wmma_kernel<8, false><<<512, 256, lds_n128, stream>>>(
        query_h, W_attn, b_attn, attn, nullptr, M_q);

    // 3) sampling offsets (N=256): f16 in, f32 out (raw offsets -> loc buffer)
    gemm_wmma_kernel<16, false><<<512, 256, lds_n256, stream>>>(
        query_h, W_off, b_off, loc, nullptr, M_q);

    // 4) softmax + location computation (in place)
    softmax_loc_kernel<<<(BATCH * LEN_Q * N_HEADS) / 256, 256, 0, stream>>>(
        attn, loc, refpt);

    // 5) bilinear sampling + weighted accumulation -> f16 (overwrites input_h region)
    sampling_kernel<<<(BATCH * LEN_Q * N_HEADS) / 8, 256, 0, stream>>>(
        value_h, attn, loc, samp_h);

    // 6) output projection: f16 in, f32 out
    gemm_wmma_kernel<16, false><<<512, 256, lds_n256, stream>>>(
        samp_h, W_out, b_out, out, nullptr, M_q);
}
